// MemoryPlus_14654428414681
// MI455X (gfx1250) — compile-verified, matrix operand order checked by reading, not run on hardware
//
#include <hip/hip_runtime.h>
#include <hip/hip_bf16.h>

#define DMODEL 1024
#define NMEM   32768
#define DKEY   256
#define DVAL   1024
#define TOPK   32
#define NTOK   4096   // B*S = 4*1024
#define CHUNK  1024   // sims columns staged in LDS per iteration

typedef __attribute__((ext_vector_type(16))) __bf16 v16bf;
typedef __attribute__((ext_vector_type(8)))  float  v8f;

union Frag { v16bf v; uint4 q[2]; };

__device__ __forceinline__ unsigned short f2bf(float f) {
  unsigned u = __float_as_uint(f);
  u += 0x7FFFu + ((u >> 16) & 1u);           // round-to-nearest-even
  return (unsigned short)(u >> 16);
}

// Load a 16(M/N) x 32(K) bf16 fragment from a row-major [rows, ld] matrix at (r0, k0).
// Matches CDNA5 16-bit A/B WMMA layout: lanes 0-15 hold K=0..7,16..23; lanes 16-31 hold K=8..15,24..31.
__device__ __forceinline__ Frag load_frag(const unsigned short* __restrict__ p,
                                          int ld, int r0, int k0, int lane) {
  Frag f;
  int row = r0 + (lane & 15);
  int kb  = k0 + ((lane & 16) ? 8 : 0);
  const unsigned short* base = p + (size_t)row * ld + kb;
  f.q[0] = *(const uint4*)(base);
  f.q[1] = *(const uint4*)(base + 16);
  return f;
}

// ---------------------------------------------------------------------------
// Elementwise f32 -> bf16 (4 elems/thread)
__global__ void cvt_bf16(const float* __restrict__ in, unsigned short* __restrict__ out, int n4) {
  int i = blockIdx.x * blockDim.x + threadIdx.x;
  if (i < n4) {
    float4 f = ((const float4*)in)[i];
    ushort4 u;
    u.x = f2bf(f.x); u.y = f2bf(f.y); u.z = f2bf(f.z); u.w = f2bf(f.w);
    ((ushort4*)out)[i] = u;
  }
}

// Row-wise L2 normalize (f32 in) -> bf16 out. One wave per row.
__global__ void rownorm_bf16(const float* __restrict__ in, unsigned short* __restrict__ out,
                             int rows, int cols) {
  int lane = threadIdx.x & 31;
  int wave = threadIdx.x >> 5;
  int row  = blockIdx.x * (blockDim.x >> 5) + wave;
  if (row >= rows) return;
  const float* p = in + (size_t)row * cols;
  float ss = 0.f;
  for (int c = lane; c < cols; c += 32) { float v = p[c]; ss += v * v; }
  for (int off = 16; off > 0; off >>= 1) ss += __shfl_xor(ss, off, 32);
  float scale = 1.0f / fmaxf(sqrtf(ss), 1e-12f);
  unsigned short* o = out + (size_t)row * cols;
  for (int c = lane; c < cols; c += 32) o[c] = f2bf(p[c] * scale);
}

// ---------------------------------------------------------------------------
// Generic bf16 WMMA GEMM: C[M,N] = A[M,K] * Bt[N,K]^T, f32 out.
// mode 0: plain store, mode 1: SiLU. One 16x16 tile per wave, 8 waves/block.
__global__ __launch_bounds__(256) void gemm_bf16(const unsigned short* __restrict__ A,
                                                 const unsigned short* __restrict__ Bt,
                                                 float* __restrict__ C,
                                                 int M, int N, int K, int mode) {
  int lane = threadIdx.x & 31;
  int wave = threadIdx.x >> 5;
  int gw   = blockIdx.x * (blockDim.x >> 5) + wave;
  int tn   = N >> 4;
  int m0   = (gw / tn) << 4;
  int n0   = (gw % tn) << 4;
  if (m0 >= M) return;   // wave-uniform

  v8f acc = {};
  for (int k0 = 0; k0 < K; k0 += 32) {
    Frag a = load_frag(A,  K, m0, k0, lane);
    Frag b = load_frag(Bt, K, n0, k0, lane);
    acc = __builtin_amdgcn_wmma_f32_16x16x32_bf16(false, a.v, false, b.v,
                                                  (short)0, acc, false, false);
  }
  int col   = n0 + (lane & 15);
  int rbase = m0 + ((lane & 16) ? 8 : 0);
  for (int r = 0; r < 8; ++r) {
    float c = acc[r];
    if (mode == 1) c = c / (1.0f + __expf(-c));   // SiLU
    C[(size_t)(rbase + r) * N + col] = c;
  }
}

// ---------------------------------------------------------------------------
// Fused sims GEMM + per-row top-32 + softmax.
// One block = 16 tokens. 8 waves WMMA a 16xCHUNK sims slab into LDS, then
// 256 threads (16/row) scan it with private threshold-gated sorted top-32 lists.
__global__ __launch_bounds__(256) void sims_topk(const unsigned short* __restrict__ qn,
                                                 const unsigned short* __restrict__ kn,
                                                 float* __restrict__ wts,
                                                 int* __restrict__ idx) {
  extern __shared__ char smem[];
  float* sims = (float*)smem;                     // 16 x CHUNK f32 = 64KB
  int tid  = threadIdx.x;
  int lane = tid & 31;
  int wave = tid >> 5;
  int tokbase = blockIdx.x << 4;

  // Preload A fragments (16 tokens x 256 K) into registers: 8 x v16bf.
  Frag afr[8];
  for (int ks = 0; ks < 8; ++ks) afr[ks] = load_frag(qn, DKEY, tokbase, ks * 32, lane);

  float tvals[TOPK]; int tidxs[TOPK];
  for (int j = 0; j < TOPK; ++j) { tvals[j] = -3.4e38f; tidxs[j] = 0; }
  float tmin = -3.4e38f;
  int row = tid >> 4;          // token within block
  int seg = tid & 15;          // 64-col segment within chunk

  for (int c = 0; c < NMEM / CHUNK; ++c) {
    // --- GEMM phase: wave w covers columns [w*128, w*128+128) of the chunk
    for (int t = 0; t < 8; ++t) {
      int loccol = wave * 128 + t * 16;
      int n0 = c * CHUNK + loccol;

      // Prefetch next tile's B rows (one row per lane pair-half) into cache:
      // lane prefetches row (next_n0 + lane&15), 512B covered by two hints.
      {
        int pn0 = (t < 7) ? (n0 + 16)
                          : ((c + 1 < NMEM / CHUNK) ? (c + 1) * CHUNK + wave * 128 : n0);
        const unsigned short* pb = kn + (size_t)(pn0 + (lane & 15)) * DKEY;
        __builtin_prefetch(pb, 0, 1);
        __builtin_prefetch(pb + 128, 0, 1);
      }

      v8f acc = {};
      for (int ks = 0; ks < 8; ++ks) {
        Frag b = load_frag(kn, DKEY, n0, ks * 32, lane);
        acc = __builtin_amdgcn_wmma_f32_16x16x32_bf16(false, afr[ks].v, false, b.v,
                                                      (short)0, acc, false, false);
      }
      int lcol = loccol + (lane & 15);
      int lrow = (lane & 16) ? 8 : 0;
      for (int r = 0; r < 8; ++r) sims[(lrow + r) * CHUNK + lcol] = acc[r];
    }
    __syncthreads();
    // --- Scan phase: threshold-gated insert into private sorted (ascending) list
    const float* srow = sims + row * CHUNK + seg * 64;
    int colbase = c * CHUNK + seg * 64;
    for (int j = 0; j < 64; ++j) {
      float v = srow[j];
      if (v > tmin) {
        int p = 0;
        while (p < TOPK - 1 && v > tvals[p + 1]) {
          tvals[p] = tvals[p + 1]; tidxs[p] = tidxs[p + 1]; ++p;
        }
        tvals[p] = v; tidxs[p] = colbase + j;
        tmin = tvals[0];
      }
    }
    __syncthreads();
  }

  // --- Merge: dump 16 sorted lists per row to LDS (reuse sims area)
  float* mv = (float*)smem;                       // 16*16*32 f32 = 32KB
  int*   mi = (int*)(smem + 32 * 1024);           // 32KB
  int lbase = (row * 16 + seg) * TOPK;
  for (int j = 0; j < TOPK; ++j) { mv[lbase + j] = tvals[j]; mi[lbase + j] = tidxs[j]; }
  __syncthreads();

  if (tid < 16) {
    int r = tid;
    int hp[16];
    for (int s = 0; s < 16; ++s) hp[s] = TOPK - 1;   // lists ascending: head = max
    float ov[TOPK]; int oi[TOPK];
    for (int o = 0; o < TOPK; ++o) {
      float best = -3.4e38f; int bs = 0;
      for (int s = 0; s < 16; ++s) {
        float v = (hp[s] >= 0) ? mv[(r * 16 + s) * TOPK + hp[s]] : -3.4e38f;
        if (v > best) { best = v; bs = s; }
      }
      ov[o] = best;
      oi[o] = mi[(r * 16 + bs) * TOPK + hp[bs]];
      hp[bs]--;
    }
    // softmax over descending ov[]
    float vmax = ov[0], sum = 0.f, w[TOPK];
    for (int o = 0; o < TOPK; ++o) { w[o] = __expf(ov[o] - vmax); sum += w[o]; }
    float inv = 1.0f / sum;
    int token = tokbase + r;
    for (int o = 0; o < TOPK; ++o) {
      wts[(size_t)token * TOPK + o] = w[o] * inv;
      idx[(size_t)token * TOPK + o] = oi[o];
    }
  }
}

// ---------------------------------------------------------------------------
// mem_out = sum_k w_k * values[idx_k]; mg = bf16(mem_out * silu_gate). One block/token.
__global__ __launch_bounds__(256) void memout_gate(const float* __restrict__ wts,
                                                   const int* __restrict__ idx,
                                                   const float* __restrict__ values,
                                                   const float* __restrict__ gate,
                                                   unsigned short* __restrict__ mg) {
  __shared__ float sw[TOPK];
  __shared__ int   si[TOPK];
  int tid = threadIdx.x;
  int token = blockIdx.x;
  if (tid < TOPK) {
    sw[tid] = wts[(size_t)token * TOPK + tid];
    si[tid] = idx[(size_t)token * TOPK + tid];
  }
  __syncthreads();
  int v0 = tid * 4;
  float4 acc = {0.f, 0.f, 0.f, 0.f};
  for (int k = 0; k < TOPK; ++k) {
    float w = sw[k];
    const float4 vv = *(const float4*)(values + (size_t)si[k] * DVAL + v0);
    acc.x += w * vv.x; acc.y += w * vv.y; acc.z += w * vv.z; acc.w += w * vv.w;
  }
  const float4 g = *(const float4*)(gate + (size_t)token * DVAL + v0);
  ushort4 o;
  o.x = f2bf(acc.x * g.x); o.y = f2bf(acc.y * g.y);
  o.z = f2bf(acc.z * g.z); o.w = f2bf(acc.w * g.w);
  *(ushort4*)(mg + (size_t)token * DVAL + v0) = o;
}

// ---------------------------------------------------------------------------
extern "C" void kernel_launch(void* const* d_in, const int* in_sizes, int n_in,
                              void* d_out, int out_size, void* d_ws, size_t ws_size,
                              hipStream_t stream) {
  const float* x      = (const float*)d_in[0];   // [4096, 1024]
  const float* keys   = (const float*)d_in[1];   // [32768, 256]
  const float* values = (const float*)d_in[2];   // [32768, 1024]
  const float* w_q    = (const float*)d_in[3];   // [256, 1024]
  const float* w_gate = (const float*)d_in[4];   // [1024, 1024]
  const float* w_out  = (const float*)d_in[5];   // [1024, 1024]
  float* out = (float*)d_out;                    // [4096, 1024]

  char* ws = (char*)d_ws;
  size_t off = 0;
  auto alloc = [&](size_t bytes) { size_t o = off; off += (bytes + 255) & ~(size_t)255; return o; };
  unsigned short* xb   = (unsigned short*)(ws + alloc((size_t)NTOK * DMODEL * 2));
  unsigned short* wqb  = (unsigned short*)(ws + alloc((size_t)DKEY * DMODEL * 2));
  unsigned short* wgb  = (unsigned short*)(ws + alloc((size_t)DVAL * DMODEL * 2));
  unsigned short* wob  = (unsigned short*)(ws + alloc((size_t)DMODEL * DVAL * 2));
  unsigned short* knb  = (unsigned short*)(ws + alloc((size_t)NMEM * DKEY * 2));
  float*          qf   = (float*)(ws + alloc((size_t)NTOK * DKEY * 4));
  unsigned short* qnb  = (unsigned short*)(ws + alloc((size_t)NTOK * DKEY * 2));
  float*          wts  = (float*)(ws + alloc((size_t)NTOK * TOPK * 4));
  int*            idxb = (int*)(ws + alloc((size_t)NTOK * TOPK * 4));
  float*          gate = (float*)(ws + alloc((size_t)NTOK * DVAL * 4));
  unsigned short* mg   = (unsigned short*)(ws + alloc((size_t)NTOK * DVAL * 2));
  (void)ws_size; (void)in_sizes; (void)n_in; (void)out_size;

  // 1) bf16 conversions
  cvt_bf16<<<(NTOK * DMODEL / 4) / 256, 256, 0, stream>>>(x, xb, NTOK * DMODEL / 4);
  cvt_bf16<<<(DKEY * DMODEL / 4) / 256, 256, 0, stream>>>(w_q, wqb, DKEY * DMODEL / 4);
  cvt_bf16<<<(DVAL * DMODEL / 4) / 256, 256, 0, stream>>>(w_gate, wgb, DVAL * DMODEL / 4);
  cvt_bf16<<<(DMODEL * DVAL / 4) / 256, 256, 0, stream>>>(w_out, wob, DMODEL * DVAL / 4);

  // 2) normalize keys -> bf16
  rownorm_bf16<<<NMEM / 8, 256, 0, stream>>>(keys, knb, NMEM, DKEY);

  // 3) q = x @ w_q^T (WMMA), then normalize -> bf16
  gemm_bf16<<<(NTOK / 16) * (DKEY / 16) / 8, 256, 0, stream>>>(xb, wqb, qf, NTOK, DKEY, DMODEL, 0);
  rownorm_bf16<<<NTOK / 8, 256, 0, stream>>>(qf, qnb, NTOK, DKEY);

  // 4) gate = silu(x @ w_gate^T) (WMMA, fused SiLU)
  gemm_bf16<<<(NTOK / 16) * (DVAL / 16) / 8, 256, 0, stream>>>(xb, wgb, gate, NTOK, DVAL, DMODEL, 1);

  // 5) fused sims GEMM + top-32 + softmax (64KB dynamic LDS)
  sims_topk<<<NTOK / 16, 256, 65536, stream>>>(qnb, knb, wts, idxb);

  // 6) weighted gather of values (L2-resident) * gate -> bf16
  memout_gate<<<NTOK, 256, 0, stream>>>(wts, idxb, values, gate, mg);

  // 7) out = (mem_out*gate) @ w_out^T (WMMA) -> f32 d_out
  gemm_bf16<<<(NTOK / 16) * (DMODEL / 16) / 8, 256, 0, stream>>>(mg, wob, out, NTOK, DMODEL, DVAL, 0);
}
